// Multitask_PS_LSTM_84988812853642
// MI455X (gfx1250) — matrix operand order repaired and checked by reading, not compiled, not run on hardware
//
#include <hip/hip_runtime.h>
#include <hip/hip_fp16.h>
#include <hip/hip_bf16.h>

typedef __attribute__((ext_vector_type(16))) _Float16     v16h;
typedef __attribute__((ext_vector_type(8)))  float        v8f;
typedef __attribute__((ext_vector_type(4)))  unsigned int u32x4;

#define BATCH 4096
#define OBS   16
#define TSTEP 32
#define HDIM  512
#define NKT   17                 // K tiles: 544 / 32  (512 h + 4 x + 1 bias + pad)
#define NJT   32                 // hidden j tiles: 512 / 16
#define HSTR  552                // padded h row stride (halves); 552*2B is 16B aligned, banks spread
#define HBUF  (16*HSTR)          // one h buffer (halves)
#define CELL_HALFS (128*NKT*512) // packed per-cell weight halves (128 n-tiles)
#define HEAD_HALFS (NKT*512)     // packed per-head weight halves (1 n-tile)
// 4 h buffers (f16) + 2 c buffers (f16, D-fragment layout) + small f32 scratch
#define SMEM_BYTES ((size_t)4*HBUF*2 + (size_t)2*8192*2 + 160*4)   // ~104 KB -> 3 blocks/WGP

#if defined(__has_builtin)
#if __has_builtin(__builtin_amdgcn_sched_group_barrier)
#define SCHED_GROUP(mask, size) __builtin_amdgcn_sched_group_barrier((mask), (size), 0)
#endif
#endif
#ifndef SCHED_GROUP
#define SCHED_GROUP(mask, size)
#endif

// ---------------- weight packing: f32 -> f16 WMMA B-fragment order ----------------
// Packed element flat index: ((ntile*NKT + kt)*512) + lane*16 + e
// maps to B[k][n] with n = ntile*16 + (lane&15), k = kt*32 + (lane&16 ? 16:0) + e
// Rows k<512 -> Wh[n][k]; k in [512,516) -> Wi[n][k-512]; k==516 -> b[n]; else 0.
__global__ void pack_cell(const float* __restrict__ Wh, const float* __restrict__ Wi,
                          const float* __restrict__ b, _Float16* __restrict__ dst)
{
    int idx = blockIdx.x * blockDim.x + threadIdx.x;
    if (idx >= CELL_HALFS) return;
    int e    = idx & 15;
    int lane = (idx >> 4) & 31;
    int tile = idx >> 9;
    int kt    = tile % NKT;
    int ntile = tile / NKT;
    int n = ntile * 16 + (lane & 15);
    int k = kt * 32 + ((lane & 16) ? 16 : 0) + e;
    float v = 0.f;
    if      (k < 512) v = Wh[(size_t)n * 512 + k];
    else if (k < 516) v = Wi[n * 4 + (k - 512)];
    else if (k == 516) v = b[n];
    dst[idx] = (_Float16)v;
}

__global__ void pack_head(const float* __restrict__ W, const float* __restrict__ b,
                          int nvalid, _Float16* __restrict__ dst)
{
    int idx = blockIdx.x * blockDim.x + threadIdx.x;
    if (idx >= HEAD_HALFS) return;
    int e    = idx & 15;
    int lane = (idx >> 4) & 31;
    int kt   = idx >> 9;
    int n = lane & 15;
    int k = kt * 32 + ((lane & 16) ? 16 : 0) + e;
    float v = 0.f;
    if (n < nvalid) {
        if      (k < 512)  v = W[(size_t)n * 512 + k];
        else if (k == 516) v = b[n];
    }
    dst[idx] = (_Float16)v;
}

// ---------------- WMMA helpers ----------------
union Frag { u32x4 q[2]; v16h v; };

// Four 16x16 gate tiles (same A, 4 B streams). sched_group_barrier keeps the clean
// [8 VMEM-read clause][4 WMMA] cadence; global_prefetch (2 k-tiles of lead, which
// flows seamlessly across jt/gate boundaries because per-gate tiles are contiguous)
// warms WGP$/L0 so the demand clause resolves at near-cache latency instead of L2.
__device__ __forceinline__ void quad_gemm(const _Float16* __restrict__ hbuf,
                                          const _Float16* __restrict__ wt0,
                                          size_t gstride, v8f* acc, int lane)
{
    const int m  = lane & 15;
    const int kb = (lane & 16) ? 16 : 0;
    const _Float16* wp[4];
#pragma unroll
    for (int g = 0; g < 4; ++g) wp[g] = wt0 + (size_t)g * gstride + lane * 16;
    const v8f vzero = {0.f,0.f,0.f,0.f,0.f,0.f,0.f,0.f};
#pragma unroll
    for (int g = 0; g < 4; ++g) acc[g] = vzero;

    Frag b[2][4];
#pragma unroll
    for (int g = 0; g < 4; ++g) {
        b[0][g].q[0] = *(const u32x4*)(wp[g]);
        b[0][g].q[1] = *(const u32x4*)(wp[g] + 8);
    }
#pragma unroll
    for (int kt = 0; kt < NKT; ++kt) {
        const int cu = kt & 1, nx = cu ^ 1;
        if (kt + 1 < NKT) {
#pragma unroll
            for (int g = 0; g < 4; ++g) {
                b[nx][g].q[0] = *(const u32x4*)(wp[g] + (kt + 1) * 512);
                b[nx][g].q[1] = *(const u32x4*)(wp[g] + (kt + 1) * 512 + 8);
            }
        }
        // Prefetch the stream 2 k-tiles ahead (wraps into the next jt tile of the
        // same gate because tiles are contiguous per gate; always inside d_ws).
#pragma unroll
        for (int g = 0; g < 4; ++g)
            __builtin_prefetch((const void*)(wp[g] + (kt + 2) * 512), 0, 3);
        Frag a;
        const _Float16* ha = hbuf + m * HSTR + kt * 32 + kb;   // 16B aligned LDS
        a.q[0] = *(const u32x4*)(ha);                          // k = base..base+7
        a.q[1] = *(const u32x4*)(ha + 8);                      // k = base+8..+15
#pragma unroll
        for (int g = 0; g < 4; ++g)
            acc[g] = __builtin_amdgcn_wmma_f32_16x16x32_f16(false, a.v, false, b[cu][g].v,
                                                            (short)0, acc[g], false, false);
        // Pin schedule: demand clause (and prefetches) ahead of the 4 WMMA.
        if (kt + 1 < NKT) SCHED_GROUP(0x020, 8);   // 8 VMEM reads (B stream)
        SCHED_GROUP(0x008, 4);                     // 4 WMMA
    }
}

// Single-tile GEMM for the small output heads (17 wmma per call).
__device__ __forceinline__ v8f gemm_tile(const _Float16* __restrict__ hbuf,
                                         const _Float16* __restrict__ wt, int lane)
{
    const int m  = lane & 15;
    const int kb = (lane & 16) ? 16 : 0;
    v8f acc = {0.f,0.f,0.f,0.f,0.f,0.f,0.f,0.f};
#pragma unroll
    for (int kt = 0; kt < NKT; ++kt) {
        Frag a, b;
        const _Float16* ha = hbuf + m * HSTR + kt * 32 + kb;
        a.q[0] = *(const u32x4*)(ha);
        a.q[1] = *(const u32x4*)(ha + 8);
        const _Float16* wpp = wt + kt * 512 + lane * 16;
        b.q[0] = *(const u32x4*)(wpp);
        b.q[1] = *(const u32x4*)(wpp + 8);
        acc = __builtin_amdgcn_wmma_f32_16x16x32_f16(false, a.v, false, b.v,
                                                     (short)0, acc, false, false);
    }
    return acc;
}

// One LSTM cell step for a 16-row tile. hcur holds [h, x, 1, 0...] rows; writes hnext.
// c kept in LDS as f16 in D-fragment layout (elementwise per-lane use only).
__device__ __forceinline__ void cell_step(const _Float16* __restrict__ hcur,
                                          _Float16* __restrict__ hnext,
                                          _Float16* __restrict__ cbuf,
                                          const _Float16* __restrict__ wcell, int lane)
{
    const int m    = lane & 15;
    const int mrow = (lane & 16) ? 8 : 0;
    for (int jt = 0; jt < NJT; ++jt) {
        v8f acc[4];   // i, f, g, o gate tiles for columns jt*16..+15
        quad_gemm(hcur, wcell + (size_t)jt * NKT * 512, (size_t)NJT * NKT * 512, acc, lane);
        _Float16* cp = cbuf + (jt * 32 + lane) * 8;   // c in D-fragment layout
        const int n = jt * 16 + m;
#pragma unroll
        for (int r = 0; r < 8; ++r) {
            float iv = 1.f / (1.f + __expf(-acc[0][r]));
            float fv = 1.f / (1.f + __expf(-acc[1][r]));
            float gv = tanhf(acc[2][r]);
            float ov = 1.f / (1.f + __expf(-acc[3][r]));
            float c  = fv * (float)cp[r] + iv * gv;
            cp[r] = (_Float16)c;
            hnext[(mrow + r) * HSTR + n] = (_Float16)(ov * tanhf(c));
        }
    }
}

// ---------------- fused persistent kernel: one wave32 per 16 batch rows ----------------
__global__ void __launch_bounds__(32)
lstm_fused(const float* __restrict__ speed, const float* __restrict__ pos,
           const _Float16* __restrict__ ws,
           const float* __restrict__ W_emb, const float* __restrict__ b_emb,
           float* __restrict__ out)
{
    extern __shared__ __align__(16) unsigned char smem[];
    _Float16* shH  = (_Float16*)smem;                       // 4 x HBUF halves
    _Float16* shC  = (_Float16*)(smem + (size_t)4 * HBUF * 2); // 2 x 8192 halves
    float*    shLs = (float*)(smem + (size_t)4 * HBUF * 2 + (size_t)2 * 8192 * 2); // 16x4
    float*    shLp = shLs + 64;                             // 16x4
    float*    shIt = shLp + 64;                             // 16x2

    const int lane = threadIdx.x;
    const int b0 = blockIdx.x * 16;

    for (int i = lane; i < 4 * HBUF; i += 32) shH[i] = (_Float16)0.f;
    for (int i = lane; i < 2 * 8192; i += 32) shC[i] = (_Float16)0.f;
    if (lane < 16)
        for (int bf = 0; bf < 4; ++bf)
            shH[bf * HBUF + lane * HSTR + 516] = (_Float16)1.f;   // bias "1" column

    // ---- encoders: enc0 -> buffers 0/1 + c[0], enc1 -> buffers 2/3 + c[1] ----
    for (int enc = 0; enc < 2; ++enc) {
        _Float16* h0 = shH + (enc * 2 + 0) * HBUF;
        _Float16* h1 = shH + (enc * 2 + 1) * HBUF;
        _Float16* cb = shC + enc * 8192;
        const _Float16* wc = ws + (size_t)enc * CELL_HALFS;
        const float* xbase = enc ? pos : speed;
        int cur = 0;
        for (int t = 0; t < OBS; ++t) {
            _Float16* hc = cur ? h1 : h0;
            _Float16* hn = cur ? h0 : h1;
            if (lane < 16) {
                float4 xv = *(const float4*)(xbase + ((size_t)(b0 + lane) * OBS + t) * 4);
                hc[lane * HSTR + 512] = (_Float16)xv.x;
                hc[lane * HSTR + 513] = (_Float16)xv.y;
                hc[lane * HSTR + 514] = (_Float16)xv.z;
                hc[lane * HSTR + 515] = (_Float16)xv.w;
            }
            cell_step(hc, hn, cb, wc, lane);
            cur ^= 1;
        }
    }
    // both encoders end with final h in buffer {0,2} (16 toggles -> back to buf 0)

    // ---- combine: h_s = h_se + h_pe (into bufs 0 and 2); c_s = c_se + c_pe ----
    for (int i = lane; i < 16 * 512; i += 32) {
        int r = i >> 9, c = i & 511;
        float v = (float)shH[0 * HBUF + r * HSTR + c] + (float)shH[2 * HBUF + r * HSTR + c];
        shH[0 * HBUF + r * HSTR + c] = (_Float16)v;
        shH[2 * HBUF + r * HSTR + c] = (_Float16)v;
    }
    for (int i = lane; i < 8192; i += 32) {
        float v = (float)shC[i] + (float)shC[8192 + i];
        shC[i] = (_Float16)v; shC[8192 + i] = (_Float16)v;
    }
    if (lane < 16) {
        float4 lsv = *(const float4*)(speed + ((size_t)(b0 + lane) * OBS + (OBS - 1)) * 4);
        float4 lpv = *(const float4*)(pos   + ((size_t)(b0 + lane) * OBS + (OBS - 1)) * 4);
        *(float4*)(shLs + lane * 4) = lsv;
        *(float4*)(shLp + lane * 4) = lpv;
    }

    // ---- decoder: 32 steps, two cells + heads ----
    const _Float16* wS  = ws + (size_t)2 * CELL_HALFS;
    const _Float16* wI  = ws + (size_t)3 * CELL_HALFS;
    const _Float16* wFs = ws + (size_t)4 * CELL_HALFS;
    const _Float16* wFc = wFs + HEAD_HALFS;
    int curS = 0, curI = 0;
    const int m    = lane & 15;
    const int mrow = (lane & 16) ? 8 : 0;
    for (int t = 0; t < TSTEP; ++t) {
        _Float16* hsC = shH + curS * HBUF;
        _Float16* hsN = shH + (curS ^ 1) * HBUF;
        _Float16* hiC = shH + (2 + curI) * HBUF;
        _Float16* hiN = shH + (2 + (curI ^ 1)) * HBUF;
        if (lane < 16) {
            for (int d = 0; d < 4; ++d) {
                hsC[lane * HSTR + 512 + d] = (_Float16)shLs[lane * 4 + d];
                hiC[lane * HSTR + 512 + d] = (_Float16)shLp[lane * 4 + d];
            }
        }
        cell_step(hsC, hsN, shC,        wS, lane);
        cell_step(hiC, hiN, shC + 8192, wI, lane);
        curS ^= 1; curI ^= 1;

        // heads via WMMA (K=544; packed zeros over x slots, bias at k=516)
        v8f sp  = gemm_tile(shH + curS * HBUF,       wFs, lane);
        v8f itv = gemm_tile(shH + (2 + curI) * HBUF, wFc, lane);

        if (m < 4) {
#pragma unroll
            for (int r = 0; r < 8; ++r) {
                float v = fminf(100.f, fmaxf(-100.f, sp[r]));
                out[(((size_t)(b0 + mrow + r)) * TSTEP + t) * 4 + m] = v;
                shLs[(mrow + r) * 4 + m] = v;                  // feedback ls
            }
        }
        if (m < 2) {
#pragma unroll
            for (int r = 0; r < 8; ++r)
                shIt[(mrow + r) * 2 + m] = fmaxf(0.f, itv[r]); // it = relu(...)
        }
        if (lane < 16) {
            float i0 = shIt[lane * 2 + 0], i1 = shIt[lane * 2 + 1];
#pragma unroll
            for (int e = 0; e < 4; ++e) {
                float v = i0 * W_emb[e * 2 + 0] + i1 * W_emb[e * 2 + 1] + b_emb[e];
                shLp[lane * 4 + e] = fmaxf(v, 0.f);            // feedback lp
            }
            if (t == TSTEP - 1) {                              // softmax of last it
                float mx = fmaxf(i0, i1);
                float e0 = __expf(i0 - mx), e1 = __expf(i1 - mx);
                float s = e0 + e1;
                out[(size_t)BATCH * TSTEP * 4 + (size_t)(b0 + lane) * 2 + 0] = e0 / s;
                out[(size_t)BATCH * TSTEP * 4 + (size_t)(b0 + lane) * 2 + 1] = e1 / s;
            }
        }
    }
}

extern "C" void kernel_launch(void* const* d_in, const int* in_sizes, int n_in,
                              void* d_out, int out_size, void* d_ws, size_t ws_size,
                              hipStream_t stream)
{
    const float* speed = (const float*)d_in[0];
    const float* pos   = (const float*)d_in[1];
    _Float16* ws = (_Float16*)d_ws;   // needs ~8.6 MB

    const int cellBlocks = (CELL_HALFS + 255) / 256;
    pack_cell<<<cellBlocks, 256, 0, stream>>>((const float*)d_in[3],  (const float*)d_in[2],
                                              (const float*)d_in[4],  ws + (size_t)0 * CELL_HALFS);
    pack_cell<<<cellBlocks, 256, 0, stream>>>((const float*)d_in[6],  (const float*)d_in[5],
                                              (const float*)d_in[7],  ws + (size_t)1 * CELL_HALFS);
    pack_cell<<<cellBlocks, 256, 0, stream>>>((const float*)d_in[9],  (const float*)d_in[8],
                                              (const float*)d_in[10], ws + (size_t)2 * CELL_HALFS);
    pack_cell<<<cellBlocks, 256, 0, stream>>>((const float*)d_in[12], (const float*)d_in[11],
                                              (const float*)d_in[13], ws + (size_t)3 * CELL_HALFS);
    const int headBlocks = (HEAD_HALFS + 255) / 256;
    pack_head<<<headBlocks, 256, 0, stream>>>((const float*)d_in[14], (const float*)d_in[15], 4,
                                              ws + (size_t)4 * CELL_HALFS);
    pack_head<<<headBlocks, 256, 0, stream>>>((const float*)d_in[16], (const float*)d_in[17], 2,
                                              ws + (size_t)4 * CELL_HALFS + HEAD_HALFS);

    lstm_fused<<<dim3(BATCH / 16), dim3(32), SMEM_BYTES, stream>>>(
        speed, pos, ws, (const float*)d_in[18], (const float*)d_in[19], (float*)d_out);
}
